// DAWN_21157008900531
// MI455X (gfx1250) — matrix operand order repaired and checked by reading, not compiled, not run on hardware
//
#include <hip/hip_runtime.h>
#include <hip/hip_bf16.h>

// ---------------- constants (match reference) ----------------
#define Bc   2
#define Sc   2048
#define Dc   1024
#define Hc   16
#define Rc   256
#define NCc  16
#define NEc  16
#define NKc  4096
#define KRc  256
#define TOPKc 8
#define DHc  64   // Dc / Hc

typedef __attribute__((ext_vector_type(16))) __bf16 v16bf;
typedef __attribute__((ext_vector_type(8)))  float  v8f;

union FragU { v16bf v; unsigned u[8]; };

#if __has_builtin(__builtin_amdgcn_tensor_load_to_lds)
#define USE_TDM 1
typedef __attribute__((ext_vector_type(4))) unsigned tdm_v4u;
typedef __attribute__((ext_vector_type(8))) int      tdm_v8i;
typedef __attribute__((ext_vector_type(4))) int      tdm_v4i;
#else
#define USE_TDM 0
#endif

static __device__ __forceinline__ unsigned short f2bf(float f) {
    union { float f; unsigned u; } x; x.f = f;
    unsigned r = x.u + 0x7fffu + ((x.u >> 16) & 1u);
    return (unsigned short)(r >> 16);
}

static __device__ __forceinline__ v8f wmma_bf16(v16bf a, v16bf b, v8f c) {
    return __builtin_amdgcn_wmma_f32_16x16x32_bf16(false, a, false, b, (short)0, c, false, false);
}

#if USE_TDM
// Issue one TDM 2D tile load: global (row-major, 2-byte elems, row stride
// `stride_elems`) -> contiguous LDS tile of tile_d1 rows x tile_d0 elems.
// Descriptor packed per CDNA5 ISA (Group0/Group1 bitfields).
static __device__ __forceinline__ void tdm_load_2d(const unsigned short* gptr,
                                                   unsigned lds_byte_off,
                                                   unsigned tile_d0, unsigned tile_d1,
                                                   unsigned long long stride_elems) {
    unsigned long long ga = (unsigned long long)(size_t)gptr;
    tdm_v4u g0;
    g0[0] = 1u;                                          // count=1, user mode
    g0[1] = lds_byte_off;                                // lds_addr
    g0[2] = (unsigned)(ga & 0xffffffffu);                // global_addr[31:0]
    g0[3] = (unsigned)((ga >> 32) & 0x1ffffffu) | (2u << 30);  // addr[56:32] | type=2
    tdm_v8i g1;
    g1[0] = (int)(1u << 16);                             // data_size=1 (2 bytes)
    g1[1] = (int)((tile_d0 & 0xffffu) << 16);            // tensor_dim0[15:0]
    g1[2] = (int)(((tile_d0 >> 16) & 0xffffu) | ((tile_d1 & 0xffffu) << 16));
    g1[3] = (int)(((tile_d1 >> 16) & 0xffffu) | ((tile_d0 & 0xffffu) << 16)); // tile_dim0
    g1[4] = (int)(tile_d1 & 0xffffu);                    // tile_dim1 (tile_dim2=0)
    g1[5] = (int)(stride_elems & 0xffffffffull);         // tensor_dim0_stride lo
    g1[6] = (int)((stride_elems >> 32) & 0xffffull);     // tensor_dim0_stride hi
    g1[7] = 0;
    tdm_v4i z = {0, 0, 0, 0};
#if __clang_major__ >= 23
    tdm_v8i z8 = {0, 0, 0, 0, 0, 0, 0, 0};
    __builtin_amdgcn_tensor_load_to_lds(g0, g1, z, z, z8, 0);
#else
    __builtin_amdgcn_tensor_load_to_lds(g0, g1, z, z, 0);
#endif
}
static __device__ __forceinline__ unsigned lds_off(const void* p) {
    return (unsigned)(size_t)p;   // flat LDS address: low 32 bits = LDS byte offset
}
#endif

// ---------------- utility kernels ----------------
__global__ void zero_f32_kernel(float* p, int n) {
    int i = blockIdx.x * 256 + threadIdx.x;
    if (i < n) p[i] = 0.f;
}

// out[c][r] = bf16(in[r][c]); in is [Rr, Cc] row-major, out is [Cc, Rr]
__global__ __launch_bounds__(256)
void transpose_f32_bf16_kernel(const float* __restrict__ in, unsigned short* __restrict__ out,
                               int Rr, int Cc) {
    __shared__ float tile[32][33];
    int c0 = blockIdx.x * 32, r0 = blockIdx.y * 32;
    int tx = threadIdx.x & 31, ty = threadIdx.x >> 5;
    for (int rr = ty; rr < 32; rr += 8)
        tile[rr][tx] = in[(size_t)(r0 + rr) * Cc + c0 + tx];
    __syncthreads();
    for (int rr = ty; rr < 32; rr += 8)
        out[(size_t)(c0 + rr) * Rr + r0 + tx] = f2bf(tile[tx][rr]);
}

// ---------------- LayerNorm + routing (one block = one token) ----------------
__global__ __launch_bounds__(256)
void ln_route_kernel(const float* __restrict__ xin, const float* __restrict__ imp,
                     const float* __restrict__ gg, const float* __restrict__ bb,
                     const float* __restrict__ W0, const float* __restrict__ W1,
                     const float* __restrict__ W2, const float* __restrict__ W3,
                     int nR, unsigned short* __restrict__ nxout, float* __restrict__ acc) {
    __shared__ float xs[Dc];
    __shared__ float red[256];
    __shared__ float red2[256];
    __shared__ float stats[2];
    __shared__ float lg[64];
    int t = blockIdx.x, tid = threadIdx.x;
    int b = t / Sc;
    const float* xrow = xin + (size_t)t * Dc;
    for (int i = tid; i < Dc; i += 256) xs[i] = xrow[i];
    __syncthreads();
    float s0 = 0.f, s1 = 0.f;
    for (int i = tid; i < Dc; i += 256) { float v = xs[i]; s0 += v; s1 += v * v; }
    red[tid] = s0; red2[tid] = s1; __syncthreads();
    for (int w = 128; w > 0; w >>= 1) {
        if (tid < w) { red[tid] += red[tid + w]; red2[tid] += red2[tid + w]; }
        __syncthreads();
    }
    if (tid == 0) {
        float mu = red[0] / Dc;
        float var = red2[0] / Dc - mu * mu;
        stats[0] = mu; stats[1] = rsqrtf(var + 1e-5f);
    }
    __syncthreads();
    float mu = stats[0], rs = stats[1];
    for (int i = tid; i < Dc; i += 256) {
        float nv = (xs[i] - mu) * rs * gg[i] + bb[i];
        xs[i] = nv;
        nxout[(size_t)t * Dc + i] = f2bf(nv);
    }
    __syncthreads();
    // router logits: P = nR*16 (expert) dot products of length Dc, 4 threads each
    int P = nR * 16;
    int p = tid >> 2, q = tid & 3;
    float part = 0.f;
    if (p < P) {
        const float* W = ((p >> 4) == 0) ? W0 : ((p >> 4) == 1) ? W1 : ((p >> 4) == 2) ? W2 : W3;
        const float* wr = W + (size_t)(p & 15) * Dc;
        for (int j = q; j < Dc; j += 4) part += xs[j] * wr[j];
    }
    red[tid] = part; __syncthreads();
    if (tid < P) lg[tid] = red[tid * 4] + red[tid * 4 + 1] + red[tid * 4 + 2] + red[tid * 4 + 3];
    __syncthreads();
    if (tid < nR) {
        int r = tid;
        float m = -1e30f;
        for (int n = 0; n < 16; n++) m = fmaxf(m, lg[r * 16 + n]);
        float e[16], sum = 0.f;
        for (int n = 0; n < 16; n++) { e[n] = __expf(lg[r * 16 + n] - m); sum += e[n]; }
        float iw = imp[t] / sum;
        for (int n = 0; n < 16; n++)
            atomicAdd(&acc[((size_t)r * Bc + b) * 16 + n], iw * e[n]);
    }
}

__global__ void route_norm_kernel(float* acc, int nRB) {
    int i = threadIdx.x;
    if (i < nRB) {
        float* w = acc + (size_t)i * 16;
        float s = 0.f;
        for (int n = 0; n < 16; n++) s += w[n];
        float inv = 1.f / (s + 1e-8f);
        for (int n = 0; n < 16; n++) w[n] *= inv;
    }
}

// out[b][i] = bf16( sum_n w[b][n] * pool[n][i] ), i < P
__global__ __launch_bounds__(256)
void mix_kernel(const float* __restrict__ w, const float* __restrict__ pool,
                unsigned short* __restrict__ out, int P) {
    int i = blockIdx.x * 256 + threadIdx.x;
    int b = blockIdx.y;
    const float* wb = w + b * 16;
    float a = 0.f;
#pragma unroll
    for (int n = 0; n < 16; n++) a += wb[n] * pool[(size_t)n * P + i];
    out[(size_t)b * P + i] = f2bf(a);
}

// ---------------- tiled WMMA GEMM (bf16 x bf16 -> f32/bf16) ----------------
// block tile 128x64, BK=32, 8 waves in 4x2 grid, each wave 32x32 (2x2 WMMA tiles)
// TDM path: double-buffered LDS tiles filled by tensor_load_to_lds (TENSORcnt),
// wave 0 drives the DMA pipeline two stages deep.
#define GBM 128
#define GBN 64
#define GBK 32
template <bool OUTF32>
__global__ __launch_bounds__(256)
void gemm_bf16_kernel(const unsigned short* __restrict__ A, const unsigned short* __restrict__ Bm,
                      void* __restrict__ Cp, int M, int N, int K, int lda, int ldb, int ldc,
                      long long sA, long long sB, long long sC,
                      const float* __restrict__ res, long long sRes, float scale) {
    int tid = threadIdx.x;
    int lane = tid & 31, wave = tid >> 5;
    int wm = wave >> 1, wn = wave & 1;
    int half = lane >> 4, r16 = lane & 15;
    int bz = blockIdx.z;
    const unsigned short* Ab = A + bz * sA;
    const unsigned short* Bb = Bm + bz * sB;
    int mbase = blockIdx.y * GBM;
    int nbase = blockIdx.x * GBN;
    v8f c[2][2] = {};

#if USE_TDM
    __shared__ __align__(16) unsigned short As[2][GBM][GBK];
    __shared__ __align__(16) unsigned short Bs[2][GBK][GBN];
    int nstage = K / GBK;
    if (wave == 0) {
        tdm_load_2d(Ab + (size_t)mbase * lda, lds_off(&As[0][0][0]), GBK, GBM, (unsigned long long)lda);
        tdm_load_2d(Bb + nbase,               lds_off(&Bs[0][0][0]), GBN, GBK, (unsigned long long)ldb);
        if (nstage > 1) {
            tdm_load_2d(Ab + (size_t)mbase * lda + GBK,       lds_off(&As[1][0][0]), GBK, GBM, (unsigned long long)lda);
            tdm_load_2d(Bb + (size_t)GBK * ldb + nbase,       lds_off(&Bs[1][0][0]), GBN, GBK, (unsigned long long)ldb);
        }
    }
    for (int s = 0; s < nstage; s++) {
        int cur = s & 1;
        if (wave == 0) {
            if (s + 1 < nstage) __builtin_amdgcn_s_wait_tensorcnt(2);
            else                __builtin_amdgcn_s_wait_tensorcnt(0);
        }
        __syncthreads();
        v16bf a[2], bfr[2];
#pragma unroll
        for (int mt = 0; mt < 2; mt++) {
            int row = wm * 32 + mt * 16 + r16;
            FragU af;
#pragma unroll
            for (int v = 0; v < 8; v++) {
                int kk = ((v >> 2) << 4) + half * 8 + ((v & 3) << 1);
                af.u[v] = *(const unsigned*)&As[cur][row][kk];
            }
            a[mt] = af.v;
        }
#pragma unroll
        for (int nt = 0; nt < 2; nt++) {
            int col = wn * 32 + nt * 16 + r16;
            FragU bf;
#pragma unroll
            for (int v = 0; v < 8; v++) {
                int kk = ((v >> 2) << 4) + half * 8 + ((v & 3) << 1);
                unsigned lo = Bs[cur][kk][col];
                unsigned hi = Bs[cur][kk + 1][col];
                bf.u[v] = lo | (hi << 16);
            }
            bfr[nt] = bf.v;
        }
#pragma unroll
        for (int mt = 0; mt < 2; mt++)
#pragma unroll
            for (int nt = 0; nt < 2; nt++)
                c[mt][nt] = wmma_bf16(a[mt], bfr[nt], c[mt][nt]);
        __syncthreads();
        if (wave == 0 && s + 2 < nstage) {
            int k2 = (s + 2) * GBK;
            tdm_load_2d(Ab + (size_t)mbase * lda + k2,  lds_off(&As[cur][0][0]), GBK, GBM, (unsigned long long)lda);
            tdm_load_2d(Bb + (size_t)k2 * ldb + nbase,  lds_off(&Bs[cur][0][0]), GBN, GBK, (unsigned long long)ldb);
        }
    }
#else
    __shared__ __align__(16) unsigned short As[GBM][GBK + 4];
    __shared__ __align__(16) unsigned short Bs[GBK][GBN + 4];
    for (int k0 = 0; k0 < K; k0 += GBK) {
        if (k0 + GBK < K) {
            __builtin_prefetch(Ab + (size_t)mbase * lda + k0 + GBK, 0, 1);
            __builtin_prefetch(Bb + (size_t)(k0 + GBK) * ldb + nbase, 0, 1);
        }
#pragma unroll
        for (int i = 0; i < 8; i++) {
            int idx = i * 256 + tid;
            int ar = idx >> 4, acu = idx & 15;
            unsigned u = *(const unsigned*)(Ab + (size_t)(mbase + ar) * lda + k0 + acu * 2);
            *(unsigned*)&As[ar][acu * 2] = u;
        }
#pragma unroll
        for (int i = 0; i < 4; i++) {
            int idx = i * 256 + tid;
            int br = idx >> 5, bcu = idx & 31;
            unsigned u = *(const unsigned*)(Bb + (size_t)(k0 + br) * ldb + nbase + bcu * 2);
            *(unsigned*)&Bs[br][bcu * 2] = u;
        }
        __syncthreads();
        v16bf a[2], bfr[2];
#pragma unroll
        for (int mt = 0; mt < 2; mt++) {
            int row = wm * 32 + mt * 16 + r16;
            FragU af;
#pragma unroll
            for (int v = 0; v < 8; v++) {
                int kk = ((v >> 2) << 4) + half * 8 + ((v & 3) << 1);
                af.u[v] = *(const unsigned*)&As[row][kk];
            }
            a[mt] = af.v;
        }
#pragma unroll
        for (int nt = 0; nt < 2; nt++) {
            int col = wn * 32 + nt * 16 + r16;
            FragU bf;
#pragma unroll
            for (int v = 0; v < 8; v++) {
                int kk = ((v >> 2) << 4) + half * 8 + ((v & 3) << 1);
                unsigned lo = Bs[kk][col];
                unsigned hi = Bs[kk + 1][col];
                bf.u[v] = lo | (hi << 16);
            }
            bfr[nt] = bf.v;
        }
#pragma unroll
        for (int mt = 0; mt < 2; mt++)
#pragma unroll
            for (int nt = 0; nt < 2; nt++)
                c[mt][nt] = wmma_bf16(a[mt], bfr[nt], c[mt][nt]);
        __syncthreads();
    }
#endif

#pragma unroll
    for (int mt = 0; mt < 2; mt++)
#pragma unroll
        for (int nt = 0; nt < 2; nt++) {
            int gc = nbase + wn * 32 + nt * 16 + r16;
#pragma unroll
            for (int v = 0; v < 8; v++) {
                int gr = mbase + wm * 32 + mt * 16 + v + half * 8;
                float val = c[mt][nt][v] * scale;
                size_t oidx = (size_t)gr * ldc + gc;
                if (res) val += res[bz * sRes + oidx];
                if (OUTF32) ((float*)Cp)[bz * sC + oidx] = val;
                else ((unsigned short*)Cp)[bz * sC + oidx] = f2bf(val);
            }
        }
}

// ---------------- flash attention: one wave = one 16-row Q tile ----------------
__global__ __launch_bounds__(256)
void attn_kernel(const unsigned short* __restrict__ Qb, const unsigned short* __restrict__ Kb,
                 const unsigned short* __restrict__ Vb, unsigned short* __restrict__ ao) {
    __shared__ __align__(16) unsigned short pbuf[8][16 * 32];
    int tid = threadIdx.x, lane = tid & 31, wave = tid >> 5;
    int h = blockIdx.y, b = blockIdx.z;
    int qbase = (blockIdx.x * 8 + wave) * 16;
    int half = lane >> 4, r16 = lane & 15;

    // Q fragments (A-layout, 16x32 bf16 each, two d-halves of dh=64)
    v16bf aQ[2];
#pragma unroll
    for (int dh2 = 0; dh2 < 2; dh2++) {
        FragU af;
        const unsigned short* qp = Qb + ((size_t)b * Sc + qbase + r16) * Dc + h * DHc + dh2 * 32;
#pragma unroll
        for (int v = 0; v < 8; v++) {
            int kk = ((v >> 2) << 4) + half * 8 + ((v & 3) << 1);
            af.u[v] = *(const unsigned*)(qp + kk);
        }
        aQ[dh2] = af.v;
    }

    float mrow[8], lrow[8];
#pragma unroll
    for (int v = 0; v < 8; v++) { mrow[v] = -1e30f; lrow[v] = 0.f; }
    v8f o[4] = {};

    for (int kv = 0; kv < qbase + 16; kv += 32) {
        bool needMask = (kv + 32 > qbase);   // wave-uniform: only diagonal strips mask
        v8f sc_[2];
#pragma unroll
        for (int jt = 0; jt < 2; jt++) {
            // B = K^T fragment: col = local K-row, contraction = d
            const unsigned short* kp = Kb + ((size_t)b * Sc + kv + jt * 16 + r16) * Dc + h * DHc;
            v8f s0 = {};
#pragma unroll
            for (int dh2 = 0; dh2 < 2; dh2++) {
                FragU bf;
#pragma unroll
                for (int v = 0; v < 8; v++) {
                    int kk = ((v >> 2) << 4) + half * 8 + ((v & 3) << 1);
                    bf.u[v] = *(const unsigned*)(kp + dh2 * 32 + kk);
                }
                s0 = wmma_bf16(aQ[dh2], bf.v, s0);
            }
            sc_[jt] = s0;
        }
        // online softmax over this 16x32 score strip
        float p[2][8];
#pragma unroll
        for (int v = 0; v < 8; v++) {
            int qr = qbase + v + half * 8;
            float lm = -1e30f;
#pragma unroll
            for (int jt = 0; jt < 2; jt++) {
                float sv = sc_[jt][v] * 0.125f;      // 1/sqrt(dh)
                if (needMask) {
                    int kc = kv + jt * 16 + r16;
                    if (kc > qr) sv = -1e30f;        // causal
                }
                p[jt][v] = sv;
                lm = fmaxf(lm, sv);
            }
#pragma unroll
            for (int m = 1; m < 16; m <<= 1) lm = fmaxf(lm, __shfl_xor(lm, m, 32));
            float mnew = fmaxf(mrow[v], lm);
            float sfac = __expf(mrow[v] - mnew);
            float rsum = 0.f;
#pragma unroll
            for (int jt = 0; jt < 2; jt++) {
                float e = __expf(p[jt][v] - mnew);
                p[jt][v] = e;
                rsum += e;
            }
#pragma unroll
            for (int m = 1; m < 16; m <<= 1) rsum += __shfl_xor(rsum, m, 32);
            lrow[v] = lrow[v] * sfac + rsum;
            mrow[v] = mnew;
#pragma unroll
            for (int nt = 0; nt < 4; nt++) o[nt][v] *= sfac;
        }
        // re-layout P (C-layout f32 -> A-layout bf16) through per-wave LDS
#pragma unroll
        for (int v = 0; v < 8; v++) {
            int rr = v + half * 8;
            pbuf[wave][rr * 32 + r16]      = f2bf(p[0][v]);
            pbuf[wave][rr * 32 + 16 + r16] = f2bf(p[1][v]);
        }
        asm volatile("s_wait_dscnt 0x0" ::: "memory");
        FragU af;
#pragma unroll
        for (int v = 0; v < 8; v++) {
            int kk = ((v >> 2) << 4) + half * 8 + ((v & 3) << 1);
            af.u[v] = *(const unsigned*)&pbuf[wave][r16 * 32 + kk];
        }
        v16bf aP = af.v;
        // P @ V: contraction over the 32 kv rows
#pragma unroll
        for (int nt = 0; nt < 4; nt++) {
            FragU bf;
#pragma unroll
            for (int v = 0; v < 8; v++) {
                int kk = ((v >> 2) << 4) + half * 8 + ((v & 3) << 1);
                unsigned lo = Vb[((size_t)b * Sc + kv + kk) * Dc + h * DHc + nt * 16 + r16];
                unsigned hi = Vb[((size_t)b * Sc + kv + kk + 1) * Dc + h * DHc + nt * 16 + r16];
                bf.u[v] = lo | (hi << 16);
            }
            o[nt] = wmma_bf16(aP, bf.v, o[nt]);
        }
    }
#pragma unroll
    for (int nt = 0; nt < 4; nt++)
#pragma unroll
        for (int v = 0; v < 8; v++) {
            int s = qbase + v + half * 8;
            float val = o[nt][v] / lrow[v];
            ao[((size_t)b * Sc + s) * Dc + h * DHc + nt * 16 + r16] = f2bf(val);
        }
}

// ---------------- top-k + softmax + gather; out += mem ----------------
__global__ __launch_bounds__(256)
void topk_mem_kernel(const float* __restrict__ ms, const float* __restrict__ kV,
                     float* __restrict__ out) {
    __shared__ float sm[NKc];
    __shared__ float rv[256];
    __shared__ int   ri[256];
    __shared__ float topv[TOPKc];
    __shared__ int   topi[TOPKc];
    __shared__ float wk[TOPKc];
    int t = blockIdx.x, tid = threadIdx.x;
    const float* row = ms + (size_t)t * NKc;
    for (int i = tid; i < NKc; i += 256) sm[i] = row[i];
    __syncthreads();
    for (int k = 0; k < TOPKc; k++) {
        float bm = -1e30f; int bi = 0;
        for (int i = tid; i < NKc; i += 256) {
            float v = sm[i];
            if (v > bm) { bm = v; bi = i; }
        }
        rv[tid] = bm; ri[tid] = bi; __syncthreads();
        for (int w = 128; w > 0; w >>= 1) {
            if (tid < w && rv[tid + w] > rv[tid]) { rv[tid] = rv[tid + w]; ri[tid] = ri[tid + w]; }
            __syncthreads();
        }
        if (tid == 0) { topv[k] = rv[0]; topi[k] = ri[0]; sm[ri[0]] = -1e30f; }
        __syncthreads();
    }
    if (tid == 0) {
        float m = topv[0], s = 0.f;
        for (int k = 0; k < TOPKc; k++) { wk[k] = __expf(topv[k] - m); s += wk[k]; }
        for (int k = 0; k < TOPKc; k++) wk[k] /= s;
    }
    __syncthreads();
    for (int d = tid; d < Dc; d += 256) {
        float a = 0.f;
#pragma unroll
        for (int k = 0; k < TOPKc; k++) a += wk[k] * kV[(size_t)topi[k] * Dc + d];
        out[(size_t)t * Dc + d] += a;
    }
}

// ---------------- host launcher ----------------
extern "C" void kernel_launch(void* const* d_in, const int* in_sizes, int n_in,
                              void* d_out, int out_size, void* d_ws, size_t ws_size,
                              hipStream_t stream) {
    (void)in_sizes; (void)n_in; (void)out_size; (void)ws_size;
    const float* x   = (const float*)d_in[0];
    const float* imp = (const float*)d_in[1];
    const float* Wc  = (const float*)d_in[2];
    const float* WQ  = (const float*)d_in[3];
    const float* WK_ = (const float*)d_in[4];
    const float* WV  = (const float*)d_in[5];
    const float* Wm  = (const float*)d_in[6];
    const float* cn  = (const float*)d_in[7];   // [NC, D, R]
    const float* ep  = (const float*)d_in[8];   // [NE, R, D]
    const float* kK  = (const float*)d_in[9];   // [NK, KR]
    const float* kV  = (const float*)d_in[10];  // [NK, D]
    const float* WO  = (const float*)d_in[11];  // [D, D]
    const float* g1  = (const float*)d_in[12];
    const float* b1  = (const float*)d_in[13];
    const float* g2  = (const float*)d_in[14];
    const float* b2  = (const float*)d_in[15];
    float* out = (float*)d_out;

    char* ws = (char*)d_ws;
    size_t off = 0;
    auto alloc = [&](size_t bytes) -> char* {
        char* p = ws + off;
        off = (off + bytes + 255) & ~(size_t)255;
        return p;
    };
    unsigned short* nx  = (unsigned short*)alloc((size_t)Bc * Sc * Dc * 2);
    float*          acc = (float*)alloc((size_t)5 * Bc * 16 * 4);
    unsigned short* sc  = (unsigned short*)alloc((size_t)Bc * Dc * Rc * 2);
    unsigned short* eq  = (unsigned short*)alloc((size_t)Bc * Rc * Dc * 2);
    unsigned short* ek  = (unsigned short*)alloc((size_t)Bc * Rc * Dc * 2);
    unsigned short* ev  = (unsigned short*)alloc((size_t)Bc * Rc * Dc * 2);
    unsigned short* mc  = (unsigned short*)alloc((size_t)Bc * Dc * Rc * 2);
    unsigned short* hb  = (unsigned short*)alloc((size_t)Bc * Sc * Rc * 2);
    unsigned short* Qb  = (unsigned short*)alloc((size_t)Bc * Sc * Dc * 2);
    unsigned short* Kb  = (unsigned short*)alloc((size_t)Bc * Sc * Dc * 2);
    unsigned short* Vb  = (unsigned short*)alloc((size_t)Bc * Sc * Dc * 2);
    unsigned short* aob = (unsigned short*)alloc((size_t)Bc * Sc * Dc * 2);
    unsigned short* WOT = (unsigned short*)alloc((size_t)Dc * Dc * 2);
    unsigned short* KKT = (unsigned short*)alloc((size_t)KRc * NKc * 2);
    float*          msb = (float*)alloc((size_t)Bc * Sc * NKc * 4);

    long long SD = (long long)Sc * Dc, SR = (long long)Sc * Rc;
    long long DR = (long long)Dc * Rc, RD = (long long)Rc * Dc;

    zero_f32_kernel<<<1, 256, 0, stream>>>(acc, 5 * Bc * 16);
    transpose_f32_bf16_kernel<<<dim3(Dc / 32, Dc / 32), 256, 0, stream>>>(WO, WOT, Dc, Dc);
    transpose_f32_bf16_kernel<<<dim3(KRc / 32, NKc / 32), 256, 0, stream>>>(kK, KKT, NKc, KRc);

    // ---- attention sub-block ----
    ln_route_kernel<<<Bc * Sc, 256, 0, stream>>>(x, imp, g1, b1, Wc, WQ, WK_, WV, 4, nx, acc);
    route_norm_kernel<<<1, 64, 0, stream>>>(acc, 4 * Bc);
    mix_kernel<<<dim3(DR / 256, Bc), 256, 0, stream>>>(acc + 0 * Bc * 16, cn, sc, (int)DR);
    mix_kernel<<<dim3(RD / 256, Bc), 256, 0, stream>>>(acc + 1 * Bc * 16, ep, eq, (int)RD);
    mix_kernel<<<dim3(RD / 256, Bc), 256, 0, stream>>>(acc + 2 * Bc * 16, ep, ek, (int)RD);
    mix_kernel<<<dim3(RD / 256, Bc), 256, 0, stream>>>(acc + 3 * Bc * 16, ep, ev, (int)RD);

    // h = nx @ sc : [S,D]x[D,R] per batch
    gemm_bf16_kernel<false><<<dim3(Rc / GBN, Sc / GBM, Bc), 256, 0, stream>>>(
        nx, sc, hb, Sc, Rc, Dc, Dc, Rc, Rc, SD, DR, SR, nullptr, 0, 1.f);
    // Q/K/V = h @ e : [S,R]x[R,D] per batch
    gemm_bf16_kernel<false><<<dim3(Dc / GBN, Sc / GBM, Bc), 256, 0, stream>>>(
        hb, eq, Qb, Sc, Dc, Rc, Rc, Dc, Dc, SR, RD, SD, nullptr, 0, 1.f);
    gemm_bf16_kernel<false><<<dim3(Dc / GBN, Sc / GBM, Bc), 256, 0, stream>>>(
        hb, ek, Kb, Sc, Dc, Rc, Rc, Dc, Dc, SR, RD, SD, nullptr, 0, 1.f);
    gemm_bf16_kernel<false><<<dim3(Dc / GBN, Sc / GBM, Bc), 256, 0, stream>>>(
        hb, ev, Vb, Sc, Dc, Rc, Rc, Dc, Dc, SR, RD, SD, nullptr, 0, 1.f);

    attn_kernel<<<dim3(Sc / 128, Hc, Bc), 256, 0, stream>>>(Qb, Kb, Vb, aob);

    // out = x + ao @ W_O^T  (fp32)
    gemm_bf16_kernel<true><<<dim3(Dc / GBN, Sc / GBM, Bc), 256, 0, stream>>>(
        aob, WOT, out, Sc, Dc, Dc, Dc, Dc, Dc, SD, 0, SD, x, SD, 1.f);

    // ---- memory sub-block ----
    ln_route_kernel<<<Bc * Sc, 256, 0, stream>>>(out, imp, g2, b2, Wm, Wm, Wm, Wm, 1, nx,
                                                 acc + 4 * Bc * 16);
    route_norm_kernel<<<1, 64, 0, stream>>>(acc + 4 * Bc * 16, 1 * Bc);
    mix_kernel<<<dim3(DR / 256, Bc), 256, 0, stream>>>(acc + 4 * Bc * 16, cn, mc, (int)DR);

    // Qm = nx2 @ mc : [S,D]x[D,R]
    gemm_bf16_kernel<false><<<dim3(Rc / GBN, Sc / GBM, Bc), 256, 0, stream>>>(
        nx, mc, hb, Sc, Rc, Dc, Dc, Rc, Rc, SD, DR, SR, nullptr, 0, 1.f);
    // ms = Qm @ kK^T / sqrt(KR) : [S,R]x[R,NK]
    gemm_bf16_kernel<true><<<dim3(NKc / GBN, Sc / GBM, Bc), 256, 0, stream>>>(
        hb, KKT, msb, Sc, NKc, Rc, Rc, NKc, NKc, SR, 0, (long long)Sc * NKc, nullptr, 0, 0.0625f);

    // top-8 + softmax + gather: out += mem
    topk_mem_kernel<<<Bc * Sc, 256, 0, stream>>>(msb, kV, out);
}
